// SGRU4Rec_17085379904472
// MI455X (gfx1250) — compile-verified, matrix operand order checked by reading, not run on hardware
//
#include <hip/hip_runtime.h>
#include <hip/hip_bf16.h>

typedef __attribute__((ext_vector_type(16))) _Float16 v16h;
typedef __attribute__((ext_vector_type(8)))  _Float16 v8h;
typedef __attribute__((ext_vector_type(4)))  _Float16 v4h;
typedef __attribute__((ext_vector_type(8)))  float    v8f;

#define T_STEPS 4
#define B_SZ    256
#define L_SZ    200
#define D_SZ    128
#define D3      384
#define ROWS    (B_SZ * L_SZ)          // 51200

// workspace layout (bytes)
#define SZ_SNN   ((size_t)T_STEPS * B_SZ * L_SZ * D_SZ * 2)   // 52,428,800
#define OFF_WIH  (SZ_SNN)
#define OFF_WHH  (OFF_WIH + (size_t)D3 * D_SZ * 2)
#define OFF_DW   (OFF_WHH + (size_t)D3 * D_SZ * 2)
#define OFF_HSEL (OFF_DW  + (size_t)D_SZ * D_SZ * 2)

__device__ __forceinline__ float sigm(float x) {
  return 1.0f / (1.0f + __expf(-x));
}
__device__ __forceinline__ float tanh_fast(float x) {
  return 1.0f - 2.0f / (1.0f + __expf(2.0f * x));
}

// ---------------------------------------------------------------------------
// Convert fp32 weights to f16 (row-major [out][in] == contiguous-K order for
// the B-fragment loads).
// ---------------------------------------------------------------------------
__global__ void wconv_kernel(const float* __restrict__ wih,
                             const float* __restrict__ whh,
                             const float* __restrict__ dw,
                             _Float16* __restrict__ wih16,
                             _Float16* __restrict__ whh16,
                             _Float16* __restrict__ dw16) {
  int i = blockIdx.x * blockDim.x + threadIdx.x;
  const int nw = D3 * D_SZ;            // 49152
  if (i < nw)                wih16[i]         = (_Float16)wih[i];
  else if (i < 2 * nw)       whh16[i - nw]    = (_Float16)whh[i - nw];
  else if (i < 2 * nw + D_SZ * D_SZ)
                             dw16[i - 2 * nw] = (_Float16)dw[i - 2 * nw];
}

// ---------------------------------------------------------------------------
// Fused embedding gather + LayerNorm + analytic 4-step LIF.
// One wave (32 lanes x 4 elems) per (b,l) row; spikes stored f16.
// snn flat index: ((t*B + b)*L + l)*D + d  ==  t*ROWS*D + row*D + d.
// ---------------------------------------------------------------------------
__global__ void embed_ln_lif_kernel(const int*   __restrict__ item_seq,
                                    const float* __restrict__ emb,
                                    const float* __restrict__ gamma,
                                    const float* __restrict__ beta,
                                    _Float16*    __restrict__ snn) {
  int wave = threadIdx.x >> 5, lane = threadIdx.x & 31;
  int row = blockIdx.x * 8 + wave;
  if (row >= ROWS) return;
  int item = item_seq[row];
  float4 x4 = *(const float4*)(emb + (size_t)item * D_SZ + lane * 4);
  float s  = x4.x + x4.y + x4.z + x4.w;
  float sq = x4.x * x4.x + x4.y * x4.y + x4.z * x4.z + x4.w * x4.w;
#pragma unroll
  for (int off = 16; off >= 1; off >>= 1) {
    s  += __shfl_xor(s,  off);
    sq += __shfl_xor(sq, off);
  }
  float m    = s * (1.0f / 128.0f);
  float var  = sq * (1.0f / 128.0f) - m * m;
  float rstd = rsqrtf(var + 1e-5f);
  float4 g4 = *(const float4*)(gamma + lane * 4);
  float4 b4 = *(const float4*)(beta  + lane * 4);
  float y[4];
  y[0] = (x4.x - m) * rstd * g4.x + b4.x;
  y[1] = (x4.y - m) * rstd * g4.y + b4.y;
  y[2] = (x4.z - m) * rstd * g4.z + b4.z;
  y[3] = (x4.w - m) * rstd * g4.w + b4.w;
  float v[4] = {0.f, 0.f, 0.f, 0.f};
#pragma unroll
  for (int t = 0; t < T_STEPS; ++t) {
    v4h sp;
#pragma unroll
    for (int i = 0; i < 4; ++i) {
      v[i] = 0.5f * (v[i] + y[i]);             // v += (x - v)/tau, tau = 2
      float sv = (v[i] >= 1.0f) ? 1.0f : 0.0f; // Heaviside(v - vth)
      v[i] -= sv;                               // soft reset
      sp[i] = (_Float16)sv;
    }
    *(v4h*)(snn + (size_t)t * ROWS * D_SZ + (size_t)row * D_SZ + lane * 4) = sp;
  }
}

// ---------------------------------------------------------------------------
// GRU recurrence. 64 blocks x 768 threads (24 waves). Block owns 16 sequences
// (one t_idx). Wave (g,c): gate g in {r,z,n}, column slice c (16 dims).
// Weight B-fragments live in VGPRs for the whole L=200 loop; spike tiles are
// software-pipelined through registers (issue load l+1 while computing l).
// GEMM phase is branch-free: 8 WMMAs/wave/step, EXEC always all-1s.
// ---------------------------------------------------------------------------
__launch_bounds__(768)
__global__ void gru_kernel(const int*      __restrict__ item_seq_len,
                           const _Float16* __restrict__ snn,
                           const _Float16* __restrict__ wih16,
                           const _Float16* __restrict__ whh16,
                           float*          __restrict__ h_sel) {
  __shared__ _Float16 s_tile[16][128];
  __shared__ _Float16 h_tile[16][128];
  __shared__ float    gates[16][384];   // r | z | xn
  __shared__ float    hn_lds[16][128];  // hidden part of n gate
  __shared__ int      sel_l[16];

  int tid  = threadIdx.x;
  int lane = tid & 31;
  // wave/gate indices as provably-scalar values -> s_cbranch, no EXEC masking
  int wave = __builtin_amdgcn_readfirstlane(tid >> 5);
  int g = wave >> 3, c = wave & 7;
  int li = lane & 15, hi = lane >> 4;
  int n0 = blockIdx.x * 16;
  int b0 = (blockIdx.x & 15) * 16;

  // ---- preload this wave's weight B-fragments (stay in VGPRs) ----
  int ncol = g * 128 + c * 16 + li;     // column of W^T == row of W
  v16h bih[4], bhh[4];
#pragma unroll
  for (int kb = 0; kb < 4; ++kb) {
    int k0 = kb * 32 + hi * 8;
    v8h lo = *(const v8h*)(wih16 + (size_t)ncol * 128 + k0);
    v8h hp = *(const v8h*)(wih16 + (size_t)ncol * 128 + k0 + 16);
    v16h a;
#pragma unroll
    for (int i = 0; i < 8; ++i) { a[i] = lo[i]; a[8 + i] = hp[i]; }
    bih[kb] = a;
    lo = *(const v8h*)(whh16 + (size_t)ncol * 128 + k0);
    hp = *(const v8h*)(whh16 + (size_t)ncol * 128 + k0 + 16);
#pragma unroll
    for (int i = 0; i < 8; ++i) { a[i] = lo[i]; a[8 + i] = hp[i]; }
    bhh[kb] = a;
  }

  // staging threads (0..255): 16 rows x 16 chunks of 8 halfs
  int sr = tid >> 4, sd0 = (tid & 15) * 8;
  const _Float16* s_src = snn + (size_t)(n0 + sr) * L_SZ * D_SZ + sd0;
  v8h s_pref = {};
  if (tid < 256) {
    s_pref = *(const v8h*)s_src;                   // spikes for l = 0
#pragma unroll
    for (int i = 0; i < 8; ++i) h_tile[sr][sd0 + i] = (_Float16)0.0f;
  }
  if (tid < 16) sel_l[tid] = item_seq_len[b0 + tid] - 1;
  __syncthreads();

  for (int l = 0; l < L_SZ; ++l) {
    // ---- commit prefetched spikes, then issue next step's load ----
    if (tid < 256) {
      *(v8h*)&s_tile[sr][sd0] = s_pref;
      int lp = (l + 1 < L_SZ) ? (l + 1) : l;
      s_pref = *(const v8h*)(s_src + (size_t)lp * D_SZ);
    }
    __syncthreads();

    // ---- WMMA (branch-free, full EXEC): acc_x = s@Wih^T, acc_h = h@Whh^T ----
    v8f acc_x = {}; v8f acc_h = {};
#pragma unroll
    for (int kb = 0; kb < 4; ++kb) {
      int kbase = kb * 32 + hi * 8;
      v16h a_s, a_h;
      {
        v8h lo = *(const v8h*)&s_tile[li][kbase];
        v8h hp = *(const v8h*)&s_tile[li][kbase + 16];
#pragma unroll
        for (int i = 0; i < 8; ++i) { a_s[i] = lo[i]; a_s[8 + i] = hp[i]; }
        lo = *(const v8h*)&h_tile[li][kbase];
        hp = *(const v8h*)&h_tile[li][kbase + 16];
#pragma unroll
        for (int i = 0; i < 8; ++i) { a_h[i] = lo[i]; a_h[8 + i] = hp[i]; }
      }
      acc_x = __builtin_amdgcn_wmma_f32_16x16x32_f16(
          false, a_s, false, bih[kb], (short)0, acc_x, false, false);
      acc_h = __builtin_amdgcn_wmma_f32_16x16x32_f16(
          false, a_h, false, bhh[kb], (short)0, acc_h, false, false);
    }

    // ---- spill to LDS (C layout: row = hi*8 + v, col = li) ----
    {
      int colg = c * 16 + li;
      int rb = hi * 8;
      if (g < 2) {                       // scalar-uniform branch
#pragma unroll
        for (int v = 0; v < 8; ++v)
          gates[rb + v][g * 128 + colg] = acc_x[v] + acc_h[v];
      } else {
#pragma unroll
        for (int v = 0; v < 8; ++v) {
          gates[rb + v][256 + colg] = acc_x[v];   // xn
          hn_lds[rb + v][colg]      = acc_h[v];   // hn
        }
      }
    }
    __syncthreads();

    // ---- elementwise GRU update + capture at len-1 (512 thr x 4 dims) ----
    if (tid < 512) {
      int r = tid >> 5, d0 = (tid & 31) * 4;
      float4 gr4 = *(const float4*)&gates[r][d0];
      float4 gz4 = *(const float4*)&gates[r][128 + d0];
      float4 xn4 = *(const float4*)&gates[r][256 + d0];
      float4 hn4 = *(const float4*)&hn_lds[r][d0];
      v4h hv = *(const v4h*)&h_tile[r][d0];
      float hnew[4];
      {
        const float gr[4] = {gr4.x, gr4.y, gr4.z, gr4.w};
        const float gz[4] = {gz4.x, gz4.y, gz4.z, gz4.w};
        const float xn[4] = {xn4.x, xn4.y, xn4.z, xn4.w};
        const float hn[4] = {hn4.x, hn4.y, hn4.z, hn4.w};
        v4h ho;
#pragma unroll
        for (int i = 0; i < 4; ++i) {
          float rr = sigm(gr[i]);
          float zz = sigm(gz[i]);
          float nn = tanh_fast(xn[i] + rr * hn[i]);
          hnew[i] = (1.0f - zz) * nn + zz * (float)hv[i];
          ho[i] = (_Float16)hnew[i];
        }
        *(v4h*)&h_tile[r][d0] = ho;
      }
      if (l == sel_l[r]) {
        float4 o = {hnew[0], hnew[1], hnew[2], hnew[3]};
        *(float4*)(h_sel + (size_t)(n0 + r) * D_SZ + d0) = o;
      }
    }
    // next-iteration barrier (after s_tile commit) orders h_tile writes
    // against the following WMMA reads.
  }
}

// ---------------------------------------------------------------------------
// Mean over T + dense (WMMA) + bias. 16 blocks x 256 threads (8 waves),
// block handles 16 output rows, wave handles 16 output columns.
// ---------------------------------------------------------------------------
__launch_bounds__(256)
__global__ void dense_kernel(const float*    __restrict__ h_sel,
                             const _Float16* __restrict__ dw16,
                             const float*    __restrict__ dense_b,
                             float*          __restrict__ out) {
  __shared__ _Float16 a_tile[16][128];
  int tid  = threadIdx.x;
  int wave = __builtin_amdgcn_readfirstlane(tid >> 5);
  int lane = tid & 31;
  int li = lane & 15, hi = lane >> 4;
  int m0 = blockIdx.x * 16;

  {
    int r = tid >> 4, d0 = (tid & 15) * 8;
#pragma unroll
    for (int i = 0; i < 8; ++i) {
      int d = d0 + i;
      float s = 0.0f;
#pragma unroll
      for (int t = 0; t < T_STEPS; ++t)
        s += h_sel[(size_t)(t * B_SZ + m0 + r) * D_SZ + d];
      a_tile[r][d] = (_Float16)(0.25f * s);
    }
  }
  __syncthreads();

  int ncol = wave * 16 + li;
  v8f acc = {};
#pragma unroll
  for (int kb = 0; kb < 4; ++kb) {
    int k0 = kb * 32 + hi * 8;
    v8h lo = *(const v8h*)(dw16 + (size_t)ncol * 128 + k0);
    v8h hp = *(const v8h*)(dw16 + (size_t)ncol * 128 + k0 + 16);
    v16h b, a;
#pragma unroll
    for (int i = 0; i < 8; ++i) { b[i] = lo[i]; b[8 + i] = hp[i]; }
    lo = *(const v8h*)&a_tile[li][k0];
    hp = *(const v8h*)&a_tile[li][k0 + 16];
#pragma unroll
    for (int i = 0; i < 8; ++i) { a[i] = lo[i]; a[8 + i] = hp[i]; }
    acc = __builtin_amdgcn_wmma_f32_16x16x32_f16(
        false, a, false, b, (short)0, acc, false, false);
  }
  float bias = dense_b[ncol];
#pragma unroll
  for (int v = 0; v < 8; ++v)
    out[(size_t)(m0 + hi * 8 + v) * D_SZ + ncol] = acc[v] + bias;
}

// ---------------------------------------------------------------------------
extern "C" void kernel_launch(void* const* d_in, const int* in_sizes, int n_in,
                              void* d_out, int out_size, void* d_ws, size_t ws_size,
                              hipStream_t stream) {
  const int*   item_seq     = (const int*)d_in[0];
  const int*   item_seq_len = (const int*)d_in[1];
  const float* emb          = (const float*)d_in[2];
  const float* ln_gamma     = (const float*)d_in[3];
  const float* ln_beta      = (const float*)d_in[4];
  const float* w_ih         = (const float*)d_in[5];
  const float* w_hh         = (const float*)d_in[6];
  const float* dense_w      = (const float*)d_in[7];
  const float* dense_b      = (const float*)d_in[8];

  char* ws = (char*)d_ws;
  _Float16* snn   = (_Float16*)(ws);
  _Float16* wih16 = (_Float16*)(ws + OFF_WIH);
  _Float16* whh16 = (_Float16*)(ws + OFF_WHH);
  _Float16* dw16  = (_Float16*)(ws + OFF_DW);
  float*    h_sel = (float*)   (ws + OFF_HSEL);
  float*    out   = (float*)d_out;

  wconv_kernel<<<448, 256, 0, stream>>>(w_ih, w_hh, dense_w,
                                        wih16, whh16, dw16);
  embed_ln_lif_kernel<<<ROWS / 8, 256, 0, stream>>>(item_seq, emb,
                                                    ln_gamma, ln_beta, snn);
  gru_kernel<<<64, 768, 0, stream>>>(item_seq_len, snn, wih16, whh16, h_sel);
  dense_kernel<<<16, 256, 0, stream>>>(h_sel, dw16, dense_b, out);
}
